// NeuralTensorLayer_4526895530379
// MI455X (gfx1250) — matrix operand order repaired and checked by reading, not compiled, not run on hardware
//
#include <hip/hip_runtime.h>

#define RNUM 16
#define BNUM 4096
#define DNUM 256
#define KNUM 64

typedef __attribute__((ext_vector_type(16))) __bf16 v16bf;
typedef __attribute__((ext_vector_type(8)))  float  v8f;

__device__ __forceinline__ unsigned short f2bf(float x) {
  unsigned int u = __float_as_uint(x);
  u += 0x7FFFu + ((u >> 16) & 1u);   // round-to-nearest-even
  return (unsigned short)(u >> 16);
}

// ---------------- Stage 1a: partial sums of gathered E rows (per r, per chunk)
__global__ void s1_partial_kernel(const int* __restrict__ e1,
                                  const float* __restrict__ E,
                                  float* __restrict__ partial) {
  const int c = blockIdx.x;   // 0..31 batch chunk of 128
  const int r = blockIdx.y;   // 0..15
  const int d = threadIdx.x;  // 0..255
  const int* idx = e1 + r * BNUM + c * 128;
  float acc = 0.f;
  for (int bb = 0; bb < 128; ++bb) {
    const int row = idx[bb];
    acc += E[(size_t)row * DNUM + d];
  }
  partial[((r * 32 + c) << 8) + d] = acc;
}

// ---------------- Stage 1b: deterministic fixed-order reduce -> s1[r][d]
__global__ void s1_reduce_kernel(const float* __restrict__ partial,
                                 float* __restrict__ s1) {
  const int r = blockIdx.x;
  const int d = threadIdx.x;
  float acc = 0.f;
  for (int c = 0; c < 32; ++c) acc += partial[((r * 32 + c) << 8) + d];
  s1[(r << 8) + d] = acc;
}

// ---------------- Stage 2: s1W + fold V -> bf16 coefficient matrices A1T/A2T
// A1T[r][k][d] = bf16(V[k,d,r]);  A2T[r][k][d] = bf16(sum_dd s1[r,dd]*W[dd,d,k,r] + V[k,256+d,r])
__global__ void stage2_kernel(const float* __restrict__ s1,
                              const float* __restrict__ W,
                              const float* __restrict__ V,
                              unsigned short* __restrict__ A1T,
                              unsigned short* __restrict__ A2T) {
  __shared__ float s1L[RNUM][DNUM + 1];  // +1 pad: avoid bank conflicts on [r][dd]
  const int tid = threadIdx.x;
  for (int i = tid; i < RNUM * DNUM; i += 256) s1L[i >> 8][i & 255] = s1[i];
  __syncthreads();

  const int dl = tid >> 4;               // 0..15
  const int r  = tid & 15;               // consecutive threads -> consecutive r (W is r-innermost)
  const int d  = blockIdx.x * 16 + dl;   // 0..255
  const int k  = blockIdx.y;             // 0..63

  float acc = 0.f;
  size_t off = (size_t)d * (KNUM * RNUM) + (size_t)k * RNUM + r;  // W[dd,d,k,r], dd=0
  for (int dd = 0; dd < DNUM; ++dd) {
    acc += s1L[r][dd] * W[off];
    off += (size_t)DNUM * KNUM * RNUM;   // advance dd
  }
  const int od = (r * KNUM + k) * DNUM + d;
  A2T[od] = f2bf(acc + V[((size_t)k * (2 * DNUM) + DNUM + d) * RNUM + r]);
  A1T[od] = f2bf(V[((size_t)k * (2 * DNUM) + d) * RNUM + r]);
}

// ---------------- Stage 3: batched GEMMs + tanh + k-reduction, bf16 WMMA
__global__ void __launch_bounds__(256) stage3_kernel(
    const int* __restrict__ e1, const int* __restrict__ e2,
    const int* __restrict__ e3, const float* __restrict__ E,
    const unsigned short* __restrict__ A1T,
    const unsigned short* __restrict__ A2T,
    const float* __restrict__ bvec, const float* __restrict__ U,
    float* __restrict__ out) {
  __shared__ __align__(16) unsigned short embLds[3][32][272];  // 272 = 256 + 16 pad
  __shared__ float partP[4][32];
  __shared__ float partN[4][32];

  const int tid = threadIdx.x;
  const int r = blockIdx.y;
  const int bbase = blockIdx.x * 32;

  const int* e1p = e1 + r * BNUM + bbase;
  const int* e2p = e2 + r * BNUM + bbase;
  const int* e3p = e3 + r * BNUM + bbase;

  // Gather 3x32 embedding rows (f32) -> bf16 LDS tile
  for (int i = tid; i < 3 * 32 * 64; i += 256) {
    const int mat = i >> 11;        // /(32*64)
    const int rem = i & 2047;
    const int n = rem >> 6;         // batch item within tile
    const int c4 = rem & 63;        // float4 column
    const int* ip = (mat == 0) ? e1p : ((mat == 1) ? e2p : e3p);
    const int row = ip[n];
    const float4 f = ((const float4*)(E + (size_t)row * DNUM))[c4];
    unsigned int lo = (unsigned)f2bf(f.x) | ((unsigned)f2bf(f.y) << 16);
    unsigned int hi = (unsigned)f2bf(f.z) | ((unsigned)f2bf(f.w) << 16);
    *(uint2*)&embLds[mat][n][c4 * 4] = make_uint2(lo, hi);
  }
  __syncthreads();

  const int lane = tid & 31;
  const int wave = tid >> 5;
  const int wm = wave & 1;   // batch half of 32-tile (M)
  const int wn = wave >> 1;  // kfeat 16-tile (N), 0..3
  const int ln = lane & 15;
  const int h  = lane >> 4;

  const int kf = wn * 16 + ln;
  const unsigned short* c1 = A1T + (size_t)(r * KNUM + kf) * DNUM;
  const unsigned short* c2 = A2T + (size_t)(r * KNUM + kf) * DNUM;
  const int mrow = wm * 16 + ln;   // this lane's A-matrix row (batch item)

  union FragBF { v16bf v; uint4 q[2]; };
  v8f acc1 = {0.f,0.f,0.f,0.f,0.f,0.f,0.f,0.f};
  v8f acc2 = acc1, acc3 = acc1;

  for (int t = 0; t < 8; ++t) {
    const int dbase = t * 32;
    FragBF a1, a2, a3, b1, b2;
    // A fragment (16x32 bf16): lane ln holds K = {8h..8h+7} U {16+8h..16+8h+7}
    a1.q[0] = *(const uint4*)&embLds[0][mrow][dbase + 8 * h];
    a1.q[1] = *(const uint4*)&embLds[0][mrow][dbase + 16 + 8 * h];
    a2.q[0] = *(const uint4*)&embLds[1][mrow][dbase + 8 * h];
    a2.q[1] = *(const uint4*)&embLds[1][mrow][dbase + 16 + 8 * h];
    a3.q[0] = *(const uint4*)&embLds[2][mrow][dbase + 8 * h];
    a3.q[1] = *(const uint4*)&embLds[2][mrow][dbase + 16 + 8 * h];
    // B fragment (32x16 bf16): lane ln holds column kf, K = {8h..8h+7} U {16+8h..16+8h+7}
    b1.q[0] = *(const uint4*)(c1 + dbase + 8 * h);
    b1.q[1] = *(const uint4*)(c1 + dbase + 16 + 8 * h);
    b2.q[0] = *(const uint4*)(c2 + dbase + 8 * h);
    b2.q[1] = *(const uint4*)(c2 + dbase + 16 + 8 * h);

    acc1 = __builtin_amdgcn_wmma_f32_16x16x32_bf16(false, a1.v, false, b1.v,
                                                   (short)0, acc1, false, false);
    acc2 = __builtin_amdgcn_wmma_f32_16x16x32_bf16(false, a2.v, false, b2.v,
                                                   (short)0, acc2, false, false);
    acc3 = __builtin_amdgcn_wmma_f32_16x16x32_bf16(false, a3.v, false, b2.v,
                                                   (short)0, acc3, false, false);
  }

  const float bias = bvec[kf * RNUM + r];  // b[K,1,R]
  const float uu   = U[kf * RNUM + r];     // U[1,K,R]

  #pragma unroll
  for (int i = 0; i < 8; ++i) {
    const int m = wm * 16 + i + 8 * h;     // batch item within 32-tile (C/D layout)
    float p = tanhf(acc1[i] + acc2[i] + bias) * uu;
    float q = tanhf(acc1[i] + acc3[i] + bias) * uu;
    // deterministic tree-reduce over the 16 N-lanes (xor<16 stays in half-wave => same m)
    for (int mask = 1; mask < 16; mask <<= 1) {
      p += __shfl_xor(p, mask, 32);
      q += __shfl_xor(q, mask, 32);
    }
    if (ln == 0) { partP[wn][m] = p; partN[wn][m] = q; }
  }
  __syncthreads();

  if (tid < 32) {
    const float sp = partP[0][tid] + partP[1][tid] + partP[2][tid] + partP[3][tid];
    const float sn = partN[0][tid] + partN[1][tid] + partN[2][tid] + partN[3][tid];
    const size_t o = (size_t)r * BNUM + bbase + tid;
    out[o] = sp;                                 // score_pos, row 0 of [2, R*B]
    out[(size_t)RNUM * BNUM + o] = sn;           // score_neg, row 1
  }
}

extern "C" void kernel_launch(void* const* d_in, const int* in_sizes, int n_in,
                              void* d_out, int out_size, void* d_ws, size_t ws_size,
                              hipStream_t stream) {
  (void)in_sizes; (void)n_in; (void)out_size; (void)ws_size;
  const int*   e1 = (const int*)d_in[0];
  const int*   e2 = (const int*)d_in[1];
  const int*   e3 = (const int*)d_in[2];
  const float* E  = (const float*)d_in[3];
  const float* W  = (const float*)d_in[4];
  const float* V  = (const float*)d_in[5];
  const float* b  = (const float*)d_in[6];
  const float* U  = (const float*)d_in[7];
  float* out = (float*)d_out;

  char* ws = (char*)d_ws;
  float* s1           = (float*)ws;                          // 16 KB
  float* partial      = (float*)(ws + 16384);                // 512 KB (dead after reduce)
  unsigned short* A1T = (unsigned short*)(ws + 16384);       // reuses partial region
  unsigned short* A2T = (unsigned short*)(ws + 16384 + 524288);  // 512 KB
  // total ws usage: 16384 + 1048576 bytes

  s1_partial_kernel<<<dim3(32, RNUM), 256, 0, stream>>>(e1, E, partial);
  s1_reduce_kernel<<<RNUM, 256, 0, stream>>>(partial, s1);
  stage2_kernel<<<dim3(DNUM / 16, KNUM), 256, 0, stream>>>(s1, W, V, A1T, A2T);
  stage3_kernel<<<dim3(BNUM / 32, RNUM), 256, 0, stream>>>(e1, e2, e3, E, A1T, A2T, b, U, out);
}
